// AttentionBlock_18416819765866
// MI455X (gfx1250) — compile-verified, matrix operand order checked by reading, not run on hardware
//
#include <hip/hip_runtime.h>

// ---------------------------------------------------------------------------
// AttentionBlock for MI455X (gfx1250, wave32, WMMA).
// B=8, C=512, N=H*W=1024, heads=8, head_dim=64, groups=32.
// All GEMMs via v_wmma_f32_16x16x32_bf16 (bf16 in, f32 accumulate).
// Channel GEMMs use 32x64 register tiles per wave (8 WMMAs / K-step with
// 2 A-frags x 4 B-frags reuse); attention uses all 8 waves in both phases.
// ---------------------------------------------------------------------------

typedef __attribute__((ext_vector_type(16))) __bf16 v16bf;
typedef __attribute__((ext_vector_type(8)))  __bf16 v8bf;
typedef __attribute__((ext_vector_type(8)))  float  v8f;

#define BB     8
#define CC     512
#define NN     1024
#define NHEAD  8
#define HD     64
#define NGRP   32
#define GCH    16          // channels per group = C / GROUPS

__device__ __forceinline__ v16bf concat8(v8bf lo, v8bf hi) {
  return __builtin_shufflevector(lo, hi, 0,1,2,3,4,5,6,7,8,9,10,11,12,13,14,15);
}

// Load one 16x32 bf16 WMMA operand tile (A or B share the lane pattern on
// CDNA5): lane L holds row (L&15), K-chunks {kb..kb+7, kb+16..kb+23},
// kb = (L>=16)?8:0. base = tile origin, row-major, `stride` elements.
__device__ __forceinline__ v16bf load_frag(const __bf16* __restrict__ base, int stride) {
  const int l   = threadIdx.x & 31;
  const int row = l & 15;
  const int kb  = (l >> 4) << 3;
  const __bf16* p = base + (size_t)row * stride + kb;
  v8bf lo = *(const v8bf*)(p);
  v8bf hi = *(const v8bf*)(p + 16);
  return concat8(lo, hi);
}

__device__ __forceinline__ v8f wmma_bf16(v16bf a, v16bf b, v8f c) {
  return __builtin_amdgcn_wmma_f32_16x16x32_bf16(false, a, false, b, (short)0, c, false, false);
}

// ---------------------------------------------------------------------------
// Kernel 1: convert qkv_w / proj_w f32 -> bf16 (tiny, bandwidth trivial)
// ---------------------------------------------------------------------------
__global__ void __launch_bounds__(256)
cvt_w_kernel(const float* __restrict__ qw, const float* __restrict__ pw,
             __bf16* __restrict__ qwb, __bf16* __restrict__ pwb) {
  int i = blockIdx.x * blockDim.x + threadIdx.x;
  int stride = gridDim.x * blockDim.x;
  for (int j = i; j < 3 * CC * CC; j += stride) qwb[j] = (__bf16)qw[j];
  for (int j = i; j < CC * CC;     j += stride) pwb[j] = (__bf16)pw[j];
}

// ---------------------------------------------------------------------------
// Kernel 2: GroupNorm. One block per (batch, group); group = 16 ch x 1024.
// Writes xn in [b][n][c] layout (c contiguous) as bf16 -> B operand of QKV GEMM.
// ---------------------------------------------------------------------------
__global__ void __launch_bounds__(256)
gn_kernel(const float* __restrict__ x, const float* __restrict__ gw,
          const float* __restrict__ gb, __bf16* __restrict__ xn) {
  const int b = blockIdx.x >> 5;
  const int g = blockIdx.x & 31;
  const float* xg = x + ((size_t)b * CC + g * GCH) * NN;
  __shared__ float s1[256], s2[256];
  const int tid = threadIdx.x;

  float sum = 0.f, sq = 0.f;
  for (int i = tid; i < GCH * NN; i += 256) {
    float v = xg[i];
    sum += v; sq += v * v;
  }
  s1[tid] = sum; s2[tid] = sq;
  __syncthreads();
  for (int off = 128; off > 0; off >>= 1) {
    if (tid < off) { s1[tid] += s1[tid + off]; s2[tid] += s2[tid + off]; }
    __syncthreads();
  }
  const float inv_n = 1.0f / (GCH * NN);
  const float mean  = s1[0] * inv_n;
  const float var   = s2[0] * inv_n - mean * mean;
  const float rstd  = rsqrtf(var + 1e-5f);

  for (int i = tid; i < GCH * NN; i += 256) {
    int cl = i >> 10;           // i / N
    int n  = i & (NN - 1);
    int c  = g * GCH + cl;
    float v = (xg[i] - mean) * rstd * gw[c] + gb[c];
    xn[((size_t)b * NN + n) * CC + c] = (__bf16)v;
  }
}

// ---------------------------------------------------------------------------
// Kernel 3: QKV GEMM. qkv[o,n] = sum_c W[o,c]*xn[c,n] + bias[o], per batch.
// 32(o) x 64(n) register tile per wave: 2 A-frags x 4 B-frags -> 8 WMMAs/K-step.
// Scatter into: Q,K : [bh][n][d] (d contig)   V : [bh][d][n] (n contig)
// ---------------------------------------------------------------------------
__global__ void __launch_bounds__(256)
qkv_kernel(const __bf16* __restrict__ W, const __bf16* __restrict__ xn,
           const float* __restrict__ bias,
           __bf16* __restrict__ q, __bf16* __restrict__ k, __bf16* __restrict__ v) {
  const int wave = threadIdx.x >> 5;
  const int l    = threadIdx.x & 31;
  const int tile = blockIdx.x * 8 + wave;     // 8 * 48 * 16 wave-tiles total
  const int b    = tile / (48 * 16);
  const int rem  = tile % (48 * 16);
  const int ost  = rem / 16;                  // 32-wide o supertile
  const int nst  = rem % 16;                  // 64-wide n supertile

  const __bf16* abase = W  + (size_t)(ost * 32) * CC;
  const __bf16* bbase = xn + ((size_t)b * NN + nst * 64) * CC;

  v8f acc[2][4] = {};
  for (int kt = 0; kt < CC / 32; ++kt) {
    v16bf a0 = load_frag(abase + kt * 32, CC);
    v16bf a1 = load_frag(abase + (size_t)16 * CC + kt * 32, CC);
    #pragma unroll
    for (int j = 0; j < 4; ++j) {
      v16bf bm = load_frag(bbase + (size_t)(j * 16) * CC + kt * 32, CC);
      acc[0][j] = wmma_bf16(a0, bm, acc[0][j]);
      acc[1][j] = wmma_bf16(a1, bm, acc[1][j]);
    }
  }

  const int hi  = (l >> 4) & 1;
  const int col = l & 15;
  #pragma unroll
  for (int i = 0; i < 2; ++i) {
    #pragma unroll
    for (int j = 0; j < 4; ++j) {
      const int n = nst * 64 + j * 16 + col;
      #pragma unroll
      for (int r = 0; r < 8; ++r) {
        const int o = ost * 32 + i * 16 + r + 8 * hi;   // D: M = r + 8*(lane>=16)
        const float val = acc[i][j][r] + bias[o];
        const int which = o >> 9;                       // 0=q 1=k 2=v
        const int c     = o & (CC - 1);
        const int head  = c >> 6;
        const int d     = c & (HD - 1);
        const size_t bh = (size_t)b * NHEAD + head;
        const __bf16 bv = (__bf16)val;
        if (which == 0)      q[(bh * NN + n) * HD + d] = bv;
        else if (which == 1) k[(bh * NN + n) * HD + d] = bv;
        else                 v[(bh * HD + d) * NN + n] = bv;
      }
    }
  }
}

// ---------------------------------------------------------------------------
// Kernel 4: attention for one (bh, 16-row query tile).
//  Phase 1: S[16,1024] = (Q_tile^T K)/8; 8 waves x 8 m-tiles x 2 WMMAs -> LDS
//  Phase 2: two-pass softmax over rows; P bf16 in LDS
//  Phase 3: out[16,64] = P x V; all 8 waves: wave = (K-half, d-tile),
//           16 WMMAs each; partials combined through the dead S buffer.
//  Result scattered to ao [b][n][c] (c contiguous) for the proj GEMM.
// ---------------------------------------------------------------------------
__global__ void __launch_bounds__(256)
attn_kernel(const __bf16* __restrict__ q, const __bf16* __restrict__ k,
            const __bf16* __restrict__ v, __bf16* __restrict__ ao) {
  __shared__ float  S[16 * NN];      // 64 KB scores; reused as f32 partials
  __shared__ __bf16 P[16 * NN];      // 32 KB exp(scores)
  __shared__ float  red[256];
  __shared__ float  rowmax[16], rowsum[16];

  const int wave = threadIdx.x >> 5;
  const int l    = threadIdx.x & 31;
  const int hi   = (l >> 4) & 1;
  const int col  = l & 15;
  const int bh   = blockIdx.x >> 6;
  const int nt   = blockIdx.x & 63;

  // ---- Phase 1: scores ----
  const __bf16* qb = q + ((size_t)bh * NN + nt * 16) * HD;
  v16bf a0 = load_frag(qb, HD);          // k = 0..31 of head_dim
  v16bf a1 = load_frag(qb + 32, HD);     // k = 32..63

  for (int mt = wave; mt < 64; mt += 8) {
    const __bf16* kb = k + ((size_t)bh * NN + mt * 16) * HD;
    v16bf b0 = load_frag(kb, HD);
    v16bf b1 = load_frag(kb + 32, HD);
    v8f acc = {};
    acc = wmma_bf16(a0, b0, acc);
    acc = wmma_bf16(a1, b1, acc);
    #pragma unroll
    for (int r = 0; r < 8; ++r)
      S[(r + 8 * hi) * NN + mt * 16 + col] = acc[r] * 0.125f;  // / sqrt(64)
  }
  __syncthreads();

  // ---- Phase 2: softmax (16 threads per row, 64 elems each) ----
  const int rr  = threadIdx.x >> 4;
  const int sub = threadIdx.x & 15;
  float m = -3.4e38f;
  #pragma unroll 8
  for (int j = 0; j < 64; ++j) m = fmaxf(m, S[rr * NN + sub * 64 + j]);
  red[threadIdx.x] = m;
  __syncthreads();
  if (sub == 0) {
    float mm = red[rr * 16];
    for (int j = 1; j < 16; ++j) mm = fmaxf(mm, red[rr * 16 + j]);
    rowmax[rr] = mm;
  }
  __syncthreads();
  const float mx = rowmax[rr];
  float ssum = 0.f;
  #pragma unroll 8
  for (int j = 0; j < 64; ++j) {
    float e = __expf(S[rr * NN + sub * 64 + j] - mx);
    ssum += e;
    P[rr * NN + sub * 64 + j] = (__bf16)e;
  }
  red[threadIdx.x] = ssum;
  __syncthreads();
  if (sub == 0) {
    float s = 0.f;
    for (int j = 0; j < 16; ++j) s += red[rr * 16 + j];
    rowsum[rr] = s;
  }
  __syncthreads();   // S is dead from here; safe to reuse as partial buffer

  // ---- Phase 3: out = P x V, all 8 waves ----
  {
    const int dt   = wave & 3;         // 16-wide slice of head_dim
    const int half = wave >> 2;        // K-half: m in [half*512, half*512+512)
    const int kb2  = hi * 8;
    v8f acc = {};
    for (int kt = half * 16; kt < half * 16 + 16; ++kt) {  // 16 x (K=32)
      const int mbase = kt * 32 + kb2;
      // A frag from LDS P: row = col (l&15), K-chunks per lane half
      v8bf plo = *(const v8bf*)&P[col * NN + mbase];
      v8bf phi = *(const v8bf*)&P[col * NN + mbase + 16];
      v16bf a = concat8(plo, phi);
      // B frag from V[bh][d][m]: column d = dt*16+col, K=m contiguous
      const __bf16* vb = v + ((size_t)bh * HD + dt * 16 + col) * NN + mbase;
      v8bf blo = *(const v8bf*)vb;
      v8bf bhi = *(const v8bf*)(vb + 16);
      v16bf bm = concat8(blo, bhi);
      acc = wmma_bf16(a, bm, acc);
    }
    // dump partials: part[half][M][d] in the dead S buffer
    #pragma unroll
    for (int r = 0; r < 8; ++r) {
      const int M = r + 8 * hi;
      S[half * (16 * HD) + M * HD + dt * 16 + col] = acc[r];
    }
  }
  __syncthreads();

  // combine halves, scale by 1/rowsum, scatter to ao[b][n][c]
  {
    const int b_   = bh >> 3;
    const int head = bh & 7;
    for (int idx = threadIdx.x; idx < 16 * HD; idx += 256) {
      const int M = idx >> 6;          // local query row
      const int d = idx & (HD - 1);
      const float o = (S[idx] + S[16 * HD + idx]) / rowsum[M];
      ao[(((size_t)b_ * NN) + nt * 16 + M) * CC + head * HD + d] = (__bf16)o;
    }
  }
}

// ---------------------------------------------------------------------------
// Kernel 5: projection GEMM + bias + residual, f32 output.
// out[b,o,n] = sum_c Wp[o,c]*ao[c,n] + pb[o] + x[b,o,n]
// 32(o) x 64(n) register tile per wave, as in the QKV GEMM.
// ---------------------------------------------------------------------------
__global__ void __launch_bounds__(256)
proj_kernel(const __bf16* __restrict__ W, const __bf16* __restrict__ ao,
            const float* __restrict__ bias, const float* __restrict__ x,
            float* __restrict__ out) {
  const int wave = threadIdx.x >> 5;
  const int l    = threadIdx.x & 31;
  const int tile = blockIdx.x * 8 + wave;     // 8 * 16 * 16 wave-tiles
  const int b    = tile / (16 * 16);
  const int rem  = tile % (16 * 16);
  const int ost  = rem / 16;                  // 32-wide o supertile
  const int nst  = rem % 16;                  // 64-wide n supertile

  const __bf16* abase = W  + (size_t)(ost * 32) * CC;
  const __bf16* bbase = ao + ((size_t)b * NN + nst * 64) * CC;

  v8f acc[2][4] = {};
  for (int kt = 0; kt < CC / 32; ++kt) {
    v16bf a0 = load_frag(abase + kt * 32, CC);
    v16bf a1 = load_frag(abase + (size_t)16 * CC + kt * 32, CC);
    #pragma unroll
    for (int j = 0; j < 4; ++j) {
      v16bf bm = load_frag(bbase + (size_t)(j * 16) * CC + kt * 32, CC);
      acc[0][j] = wmma_bf16(a0, bm, acc[0][j]);
      acc[1][j] = wmma_bf16(a1, bm, acc[1][j]);
    }
  }

  const int hi  = (l >> 4) & 1;
  const int col = l & 15;
  #pragma unroll
  for (int i = 0; i < 2; ++i) {
    #pragma unroll
    for (int j = 0; j < 4; ++j) {
      const int n = nst * 64 + j * 16 + col;
      #pragma unroll
      for (int r = 0; r < 8; ++r) {
        const int o = ost * 32 + i * 16 + r + 8 * hi;
        const size_t idx = ((size_t)b * CC + o) * NN + n;
        out[idx] = acc[i][j][r] + bias[o] + x[idx];
      }
    }
  }
}

// ---------------------------------------------------------------------------
// Launch
// ---------------------------------------------------------------------------
extern "C" void kernel_launch(void* const* d_in, const int* in_sizes, int n_in,
                              void* d_out, int out_size, void* d_ws, size_t ws_size,
                              hipStream_t stream) {
  const float* x      = (const float*)d_in[0];
  const float* gn_w   = (const float*)d_in[1];
  const float* gn_b   = (const float*)d_in[2];
  const float* qkv_w  = (const float*)d_in[3];
  const float* qkv_b  = (const float*)d_in[4];
  const float* proj_w = (const float*)d_in[5];
  const float* proj_b = (const float*)d_in[6];
  float* out = (float*)d_out;

  char* ws = (char*)d_ws;
  const size_t XN_B  = (size_t)BB * NN * CC * 2;          // 8 MB
  const size_t QWB_B = (size_t)3 * CC * CC * 2;           // 1.5 MB
  const size_t PWB_B = (size_t)CC * CC * 2;               // 0.5 MB
  const size_t QKV_B = (size_t)BB * NHEAD * NN * HD * 2;  // 8 MB each

  __bf16* xn   = (__bf16*)(ws);
  __bf16* qwb  = (__bf16*)(ws + XN_B);
  __bf16* pwb  = (__bf16*)(ws + XN_B + QWB_B);
  __bf16* qbuf = (__bf16*)(ws + XN_B + QWB_B + PWB_B);
  __bf16* kbuf = (__bf16*)(ws + XN_B + QWB_B + PWB_B + QKV_B);
  __bf16* vbuf = (__bf16*)(ws + XN_B + QWB_B + PWB_B + 2 * QKV_B);
  __bf16* ao   = (__bf16*)(ws + XN_B + QWB_B + PWB_B + 3 * QKV_B);

  cvt_w_kernel<<<1024, 256, 0, stream>>>(qkv_w, proj_w, qwb, pwb);
  gn_kernel<<<BB * NGRP, 256, 0, stream>>>(x, gn_w, gn_b, xn);
  qkv_kernel<<<(BB * 48 * 16) / 8, 256, 0, stream>>>(qwb, xn, qkv_b, qbuf, kbuf, vbuf);
  attn_kernel<<<BB * NHEAD * (NN / 16), 256, 0, stream>>>(qbuf, kbuf, vbuf, ao);
  proj_kernel<<<(BB * 16 * 16) / 8, 256, 0, stream>>>(pwb, ao, proj_b, x, out);
}
